// Protrait_23656679867663
// MI455X (gfx1250) — matrix operand order repaired and checked by reading, not compiled, not run on hardware
//
#include <hip/hip_runtime.h>

#define BB   2
#define LL   1000
#define DD   288
#define HH   8
#define DIMM 36
#define DFFF 576
#define KSS  250
#define NQQ  250
#define MTOT (BB*LL)     // 2000
#define BHH  (BB*HH)     // 16
#define QPAD 256
#define KPAD 1008        // 63 * 16 key tiles

typedef __attribute__((ext_vector_type(16))) _Float16 v16h;
typedef __attribute__((ext_vector_type(8)))  float    v8f;

__device__ __forceinline__ v8f wmma_f32_16x16x32(v16h a, v16h b, v8f c) {
  // (neg_a, A, neg_b, B, c_mod, C, reuse_a, reuse_b)
  return __builtin_amdgcn_wmma_f32_16x16x32_f16(false, a, false, b, (short)0, c, false, false);
}

// ---- WMMA fragment loaders (wave32; contiguous k order, same for A and B) ----
// A (16x32 f16): lanes 0-15 row M=lane hold K 0..15; lanes 16-31 hold K 16..31
__device__ __forceinline__ v16h load_a_f16(const _Float16* A, int lda, int mbase,
                                           int k0, int kmax) {
  int lane = threadIdx.x & 31;
  int row  = mbase + (lane & 15);
  int khi  = (lane >> 4) << 4;          // 0 or 16
  const _Float16* p = A + (long)row * lda;
  v16h a;
#pragma unroll
  for (int e = 0; e < 16; ++e) {
    int k = k0 + khi + e;
    a[e] = (k < kmax) ? p[k] : (_Float16)0.f;
  }
  return a;
}

// B (32x16) built from row-major weight W[N][K]: B[k][n] = W[n][k]
// lanes 0-15 hold K 0..15 (col n=lane), lanes 16-31 hold K 16..31
__device__ __forceinline__ v16h load_b_wt_f16(const _Float16* W, int ldw, int nbase,
                                              int k0, int kmax) {
  int lane = threadIdx.x & 31;
  int n    = nbase + (lane & 15);
  int khi  = (lane >> 4) << 4;          // 0 or 16
  const _Float16* p = W + (long)n * ldw;
  v16h b;
#pragma unroll
  for (int e = 0; e < 16; ++e) {
    int k = k0 + khi + e;
    b[e] = (k < kmax) ? p[k] : (_Float16)0.f;
  }
  return b;
}

// ---------------- conversion ----------------
__global__ void cvt_f32_to_f16(const float* s, _Float16* d, int n) {
  int i = blockIdx.x * blockDim.x + threadIdx.x;
  if (i < n) d[i] = (_Float16)s[i];
}

// ------- QKV projection: C = Xh @ Wh^T + bias, N-blocked x2, dual f32/f16 output -------
__global__ void qkv_gemm(const _Float16* Xh, const _Float16* Wh, const float* bias,
                         float* dst, _Float16* dsth) {
  int mt  = blockIdx.x * blockDim.y + threadIdx.y;
  int nt2 = blockIdx.y;                 // pair of N tiles
  if (mt >= MTOT / 16) return;          // wave-uniform
  v8f c0 = {0.f,0.f,0.f,0.f,0.f,0.f,0.f,0.f};
  v8f c1 = c0;
  for (int k0 = 0; k0 < DD; k0 += 32) {
    __builtin_prefetch((const void*)(Xh + (long)(mt*16)*DD + k0 + 32), 0, 3);
    __builtin_prefetch((const void*)(Wh + (long)(nt2*32)*DD + k0 + 32), 0, 3);
    v16h a  = load_a_f16(Xh, DD, mt*16, k0, DD);
    v16h b0 = load_b_wt_f16(Wh, DD, nt2*32,      k0, DD);
    v16h b1 = load_b_wt_f16(Wh, DD, nt2*32 + 16, k0, DD);
    c0 = wmma_f32_16x16x32(a, b0, c0);
    c1 = wmma_f32_16x16x32(a, b1, c1);
  }
  int lane = threadIdx.x & 31;
  int moff = (lane >> 4) << 3;
#pragma unroll
  for (int j = 0; j < 2; ++j) {
    int n  = nt2*32 + j*16 + (lane & 15);
    int h  = n / DIMM, dd = n % DIMM;
    float bn = bias[n];
    const v8f& c = j ? c1 : c0;
#pragma unroll
    for (int r = 0; r < 8; ++r) {
      int m = mt*16 + moff + r;
      int b = m / LL, l = m % LL;
      long o = (((long)b*HH + h)*LL + l)*DIMM + dd;
      float v = c[r] + bn;
      dst[o]  = v;
      dsth[o] = (_Float16)v;
    }
  }
}

// ---------------- FFN GEMM: C = A @ W^T + bias, exact-GELU, N-blocked x2 ----------------
template<bool OUT_F16>
__global__ void ffn_gemm(const _Float16* Ah, int K, const _Float16* Wh,
                         const float* bias, void* dstv, int N) {
  int mt  = blockIdx.x * blockDim.y + threadIdx.y;
  int nt2 = blockIdx.y;
  if (mt >= MTOT / 16) return;
  v8f c0 = {0.f,0.f,0.f,0.f,0.f,0.f,0.f,0.f};
  v8f c1 = c0;
  for (int k0 = 0; k0 < K; k0 += 32) {
    __builtin_prefetch((const void*)(Ah + (long)(mt*16)*K + k0 + 32), 0, 3);
    __builtin_prefetch((const void*)(Wh + (long)(nt2*32)*K + k0 + 32), 0, 3);
    v16h a  = load_a_f16(Ah, K, mt*16, k0, K);
    v16h b0 = load_b_wt_f16(Wh, K, nt2*32,      k0, K);
    v16h b1 = load_b_wt_f16(Wh, K, nt2*32 + 16, k0, K);
    c0 = wmma_f32_16x16x32(a, b0, c0);
    c1 = wmma_f32_16x16x32(a, b1, c1);
  }
  int lane = threadIdx.x & 31;
  int moff = (lane >> 4) << 3;
#pragma unroll
  for (int j = 0; j < 2; ++j) {
    int n = nt2*32 + j*16 + (lane & 15);
    float bn = bias[n];
    const v8f& c = j ? c1 : c0;
#pragma unroll
    for (int r = 0; r < 8; ++r) {
      int m = mt*16 + moff + r;
      float xv = c[r] + bn;
      float g  = 0.5f * xv * (1.f + erff(xv * 0.70710678118f));   // exact GELU
      if (OUT_F16) ((_Float16*)dstv)[(long)m*N + n] = (_Float16)g;
      else         ((float*)dstv)   [(long)m*N + n] = g;
    }
  }
}

// ---------------- sampled-score sparsity measure: M(q) = max - sum/L ----------------
__global__ void measure_kernel(const float* qf, const float* kf, const int* index_key,
                               float* measure) {
  int bh   = blockIdx.y;
  int wid  = threadIdx.x >> 5;
  int lane = threadIdx.x & 31;
  int l = blockIdx.x * (blockDim.x >> 5) + wid;
  if (l >= LL) return;
  const float* qp = qf + ((long)bh*LL + l)*DIMM;
  const float* km = kf + (long)bh*LL*DIMM;
  const int*  idx = index_key + (long)l*KSS;
  float qreg[DIMM];
#pragma unroll
  for (int t = 0; t < DIMM; ++t) qreg[t] = qp[t];
  float mx = -3.4e38f, sm = 0.f;
  for (int j = lane; j < KSS; j += 32) {
    const float* kr = km + (long)idx[j]*DIMM;
    float d = 0.f;
#pragma unroll
    for (int t = 0; t < DIMM; ++t) d += qreg[t]*kr[t];
    mx = fmaxf(mx, d); sm += d;
  }
#pragma unroll
  for (int off = 16; off > 0; off >>= 1) {
    mx = fmaxf(mx, __shfl_xor(mx, off, 32));
    sm += __shfl_xor(sm, off, 32);
  }
  if (lane == 0) measure[(long)bh*LL + l] = mx - sm * (1.f/(float)LL);
}

// ---------------- top-NQ per (b,h) via 1024-element bitonic sort in LDS ----------------
__global__ void topk_kernel(const float* measure, int* idxq) {
  int bh = blockIdx.x;
  __shared__ float sv[1024];
  __shared__ int   si[1024];
  for (int i = threadIdx.x; i < 1024; i += blockDim.x) {
    sv[i] = (i < LL) ? measure[(long)bh*LL + i] : -3.4e38f;
    si[i] = i;
  }
  __syncthreads();
  for (int k = 2; k <= 1024; k <<= 1) {
    for (int j = k >> 1; j > 0; j >>= 1) {
      for (int i = threadIdx.x; i < 1024; i += blockDim.x) {
        int ixj = i ^ j;
        if (ixj > i) {
          float va = sv[i], vb = sv[ixj];
          int   ia = si[i], ib = si[ixj];
          bool aFirst = (va > vb) || (va == vb && ia < ib);   // descending
          bool wantDesc = ((i & k) == 0);
          if (wantDesc ? !aFirst : aFirst) {
            sv[i] = vb; sv[ixj] = va; si[i] = ib; si[ixj] = ia;
          }
        }
      }
      __syncthreads();
    }
  }
  for (int i = threadIdx.x; i < QPAD; i += blockDim.x)
    idxq[bh*QPAD + i] = si[i < NQQ ? i : (NQQ-1)];
}

// ---- sim = q_sel @ k^T / sqrt(dim) via WMMA (K pad 36->64), key-tile-blocked x3 ----
__global__ void sim_gemm(const _Float16* qh, const _Float16* kh, const int* idxq,
                         float* sim) {
  int grp = blockIdx.x * blockDim.y + threadIdx.y;   // group of 3 key tiles
  int qt  = blockIdx.y;
  int bh  = blockIdx.z;
  if (grp >= (KPAD/16)/3) return;                    // 21 groups, wave-uniform
  const _Float16* q  = qh + (long)bh*LL*DIMM;
  const _Float16* km = kh + (long)bh*LL*DIMM;
  const int*      iq = idxq + bh*QPAD;
  int lane = threadIdx.x & 31;
  int qrow = qt*16 + (lane & 15);
  int qi = iq[qrow < NQQ ? qrow : (NQQ-1)];
  const _Float16* qp = q + (long)qi*DIMM;
  int khiA = (lane >> 4) << 4;
  int khiB = (lane >> 4) << 4;
  const _Float16* kp[3];
  int key[3];
#pragma unroll
  for (int j = 0; j < 3; ++j) {
    key[j] = (grp*3 + j)*16 + (lane & 15);
    int keyc = key[j] < LL ? key[j] : (LL-1);
    kp[j] = km + (long)keyc*DIMM;
  }
  v8f c0 = {0.f,0.f,0.f,0.f,0.f,0.f,0.f,0.f};
  v8f c1 = c0, c2 = c0;
  for (int k0 = 0; k0 < 64; k0 += 32) {
    v16h a, b0, b1, b2;
#pragma unroll
    for (int e = 0; e < 16; ++e) {
      int ka = k0 + khiA + e;
      a[e] = (ka < DIMM) ? qp[ka] : (_Float16)0.f;
      int kb = k0 + khiB + e;
      bool ok = kb < DIMM;
      b0[e] = ok ? kp[0][kb] : (_Float16)0.f;
      b1[e] = ok ? kp[1][kb] : (_Float16)0.f;
      b2[e] = ok ? kp[2][kb] : (_Float16)0.f;
    }
    c0 = wmma_f32_16x16x32(a, b0, c0);
    c1 = wmma_f32_16x16x32(a, b1, c1);
    c2 = wmma_f32_16x16x32(a, b2, c2);
  }
  int moff = (lane >> 4) << 3;
#pragma unroll
  for (int j = 0; j < 3; ++j) {
    const v8f& c = (j == 0) ? c0 : (j == 1) ? c1 : c2;
#pragma unroll
    for (int r = 0; r < 8; ++r) {
      int m = qt*16 + moff + r;
      sim[((long)bh*QPAD + m)*KPAD + key[j]] = c[r] * (1.f/6.f);   // 1/sqrt(36)
    }
  }
}

// ---------------- row softmax over 1000 keys; writes attn f16 (padded/zeroed) ----------------
__global__ void softmax_kernel(const float* sim, _Float16* attn) {
  int bh = blockIdx.y, qrow = blockIdx.x;
  const float* srow = sim  + ((long)bh*QPAD + qrow)*KPAD;
  _Float16*    arow = attn + ((long)bh*QPAD + qrow)*KPAD;
  int tid = threadIdx.x;
  if (qrow >= NQQ) {                    // lazy/pad rows contribute zero
    for (int i = tid; i < KPAD; i += blockDim.x) arow[i] = (_Float16)0.f;
    return;
  }
  __shared__ float red[128];
  float mx = -3.4e38f;
  for (int i = tid; i < LL; i += 128) mx = fmaxf(mx, srow[i]);
  red[tid] = mx; __syncthreads();
  for (int off = 64; off > 0; off >>= 1) {
    if (tid < off) red[tid] = fmaxf(red[tid], red[tid+off]);
    __syncthreads();
  }
  float m = red[0]; __syncthreads();
  float s = 0.f;
  for (int i = tid; i < LL; i += 128) s += __expf(srow[i] - m);
  red[tid] = s; __syncthreads();
  for (int off = 64; off > 0; off >>= 1) {
    if (tid < off) red[tid] += red[tid+off];
    __syncthreads();
  }
  float inv = 1.f / red[0];
  for (int i = tid; i < LL; i += 128) arow[i] = (_Float16)(__expf(srow[i]-m)*inv);
  for (int i = LL + tid; i < KPAD; i += 128) arow[i] = (_Float16)0.f;
}

// ------- out_sel = attn @ v via WMMA (K pad 1000->1024), all 3 N tiles per wave -------
__global__ void outsel_gemm(const _Float16* attn, const _Float16* vh, float* outsel) {
  int qt = blockIdx.x * blockDim.y + threadIdx.y;
  int bh = blockIdx.y;
  if (qt >= QPAD/16) return;
  const _Float16* A  = attn + (long)bh*QPAD*KPAD;
  const _Float16* Vm = vh   + (long)bh*LL*DIMM;
  int lane = threadIdx.x & 31;
  int khiB = (lane >> 4) << 4;
  int n[3];
  bool nok[3];
#pragma unroll
  for (int j = 0; j < 3; ++j) {
    n[j]   = j*16 + (lane & 15);
    nok[j] = n[j] < DIMM;
  }
  v8f c0 = {0.f,0.f,0.f,0.f,0.f,0.f,0.f,0.f};
  v8f c1 = c0, c2 = c0;
  for (int k0 = 0; k0 < LL; k0 += 32) {
    __builtin_prefetch((const void*)(A + (long)(qt*16)*KPAD + k0 + 32), 0, 3);
    v16h a = load_a_f16(A, KPAD, qt*16, k0, LL);
    v16h b0, b1, b2;
#pragma unroll
    for (int e = 0; e < 16; ++e) {
      int k = k0 + khiB + e;
      bool kok = k < LL;
      const _Float16* vr = Vm + (long)(kok ? k : 0)*DIMM;
      b0[e] = (kok && nok[0]) ? vr[n[0]] : (_Float16)0.f;
      b1[e] = (kok && nok[1]) ? vr[n[1]] : (_Float16)0.f;
      b2[e] = (kok && nok[2]) ? vr[n[2]] : (_Float16)0.f;
    }
    c0 = wmma_f32_16x16x32(a, b0, c0);
    c1 = wmma_f32_16x16x32(a, b1, c1);
    c2 = wmma_f32_16x16x32(a, b2, c2);
  }
  int moff = (lane >> 4) << 3;
#pragma unroll
  for (int j = 0; j < 3; ++j) {
    const v8f& c = (j == 0) ? c0 : (j == 1) ? c1 : c2;
#pragma unroll
    for (int r = 0; r < 8; ++r) {
      int m = qt*16 + moff + r;
      outsel[((long)bh*QPAD + m)*48 + n[j]] = c[r];
    }
  }
}

// ---------------- mean(V) over keys ----------------
__global__ void vmean_kernel(const float* vf, float* vmean) {
  int bh = blockIdx.x, d = threadIdx.x;
  if (d >= DIMM) return;
  float s = 0.f;
  for (int l = 0; l < LL; ++l) s += vf[((long)bh*LL + l)*DIMM + d];
  vmean[bh*DIMM + d] = s * (1.f/(float)LL);
}

__global__ void fill_out_kernel(const float* vmean, float* attnout) {
  int bh = blockIdx.y, l = blockIdx.x, d = threadIdx.x;
  if (d >= DIMM) return;
  attnout[((long)bh*LL + l)*DIMM + d] = vmean[bh*DIMM + d];
}

__global__ void scatter_kernel(const int* idxq, const float* outsel, float* attnout) {
  int bh = blockIdx.y, q = blockIdx.x, d = threadIdx.x;
  if (d >= DIMM) return;
  int l = idxq[bh*QPAD + q];
  attnout[((long)bh*LL + l)*DIMM + d] = outsel[((long)bh*QPAD + q)*48 + d];
}

// ---------------- LayerNorm kernels (288 threads = one channel each) ----------------
__global__ void ln1_kernel(const float* x, const float* attnout, const float* g,
                           const float* bt, float* h1, _Float16* h1h) {
  int m = blockIdx.x;
  int b = m / LL, l = m % LL;
  int d = threadIdx.x;
  float val = x[(long)m*DD + d]
            + attnout[(((long)b*HH + d/DIMM)*LL + l)*DIMM + (d % DIMM)];
  __shared__ float red[DD];
  red[d] = val; __syncthreads();
  for (int off = 256; off > 0; off >>= 1) {
    if (d < off && d + off < DD) red[d] += red[d+off];
    __syncthreads();
  }
  float mu = red[0] * (1.f/(float)DD); __syncthreads();
  float dv = val - mu;
  red[d] = dv*dv; __syncthreads();
  for (int off = 256; off > 0; off >>= 1) {
    if (d < off && d + off < DD) red[d] += red[d+off];
    __syncthreads();
  }
  float var = red[0] * (1.f/(float)DD);
  float y = dv * rsqrtf(var + 1e-5f) * g[d] + bt[d];
  h1 [(long)m*DD + d] = y;
  h1h[(long)m*DD + d] = (_Float16)y;
}

__global__ void ln2_kernel(const float* g2, const float* h1, const float* g,
                           const float* bt, float* out) {
  int m = blockIdx.x;
  int d = threadIdx.x;
  float val = g2[(long)m*DD + d] + h1[(long)m*DD + d];
  __shared__ float red[DD];
  red[d] = val; __syncthreads();
  for (int off = 256; off > 0; off >>= 1) {
    if (d < off && d + off < DD) red[d] += red[d+off];
    __syncthreads();
  }
  float mu = red[0] * (1.f/(float)DD); __syncthreads();
  float dv = val - mu;
  red[d] = dv*dv; __syncthreads();
  for (int off = 256; off > 0; off >>= 1) {
    if (d < off && d + off < DD) red[d] += red[d+off];
    __syncthreads();
  }
  float var = red[0] * (1.f/(float)DD);
  out[(long)m*DD + d] = dv * rsqrtf(var + 1e-5f) * g[d] + bt[d];
}

// =====================================================================
extern "C" void kernel_launch(void* const* d_in, const int* in_sizes, int n_in,
                              void* d_out, int out_size, void* d_ws, size_t ws_size,
                              hipStream_t stream) {
  const float* x         = (const float*)d_in[0];
  const int*   index_key = (const int*)  d_in[1];
  const float* Wq = (const float*)d_in[2];  const float* bq = (const float*)d_in[3];
  const float* Wk = (const float*)d_in[4];  const float* bk = (const float*)d_in[5];
  const float* Wv = (const float*)d_in[6];  const float* bv = (const float*)d_in[7];
  const float* W1 = (const float*)d_in[8];  const float* b1 = (const float*)d_in[9];
  const float* W2 = (const float*)d_in[10]; const float* b2 = (const float*)d_in[11];
  const float* ln1g = (const float*)d_in[12]; const float* ln1b = (const float*)d_in[13];
  const float* ln2g = (const float*)d_in[14]; const float* ln2b = (const float*)d_in[15];
  float* out = (float*)d_out;

  char* ws = (char*)d_ws;
  size_t off = 0;
  auto alloc = [&](size_t bytes) -> void* {
    void* p = ws + off;
    off = (off + bytes + 255) & ~(size_t)255;
    return p;
  };
  _Float16* xh   = (_Float16*)alloc((size_t)MTOT*DD*2);
  _Float16* Wqh  = (_Float16*)alloc((size_t)DD*DD*2);
  _Float16* Wkh  = (_Float16*)alloc((size_t)DD*DD*2);
  _Float16* Wvh  = (_Float16*)alloc((size_t)DD*DD*2);
  _Float16* W1h  = (_Float16*)alloc((size_t)DFFF*DD*2);
  _Float16* W2h  = (_Float16*)alloc((size_t)DD*DFFF*2);
  float* qf      = (float*)alloc((size_t)BHH*LL*DIMM*4);
  float* kf      = (float*)alloc((size_t)BHH*LL*DIMM*4);
  float* vf      = (float*)alloc((size_t)BHH*LL*DIMM*4);
  _Float16* qh   = (_Float16*)alloc((size_t)BHH*LL*DIMM*2);
  _Float16* kh   = (_Float16*)alloc((size_t)BHH*LL*DIMM*2);
  _Float16* vh   = (_Float16*)alloc((size_t)BHH*LL*DIMM*2);
  float* meas    = (float*)alloc((size_t)BHH*LL*4);
  int*   idxq    = (int*)  alloc((size_t)BHH*QPAD*4);
  float* simb    = (float*)alloc((size_t)BHH*QPAD*KPAD*4);
  _Float16* attn = (_Float16*)alloc((size_t)BHH*QPAD*KPAD*2);
  float* outsel  = (float*)alloc((size_t)BHH*QPAD*48*4);
  float* vmean   = (float*)alloc((size_t)BHH*DIMM*4);
  float* attnout = (float*)alloc((size_t)BHH*LL*DIMM*4);
  float* h1      = (float*)alloc((size_t)MTOT*DD*4);
  _Float16* h1h  = (_Float16*)alloc((size_t)MTOT*DD*2);
  _Float16* f1h  = (_Float16*)alloc((size_t)MTOT*DFFF*2);
  float* g2      = (float*)alloc((size_t)MTOT*DD*4);
  (void)ws_size; (void)n_in; (void)in_sizes; (void)out_size;

  // 1) f32 -> f16 conversions (activations + weights)
  auto cvt = [&](const float* s, _Float16* d, int n) {
    cvt_f32_to_f16<<<(n + 255)/256, 256, 0, stream>>>(s, d, n);
  };
  cvt(x,  xh,  MTOT*DD);
  cvt(Wq, Wqh, DD*DD);  cvt(Wk, Wkh, DD*DD);  cvt(Wv, Wvh, DD*DD);
  cvt(W1, W1h, DFFF*DD); cvt(W2, W2h, DD*DFFF);

  // 2) QKV projections (WMMA, N-blocked x2): 125 M-tiles x 9 N-pairs
  {
    dim3 blk(32, 4), grd((MTOT/16 + 3)/4, DD/32);
    qkv_gemm<<<grd, blk, 0, stream>>>(xh, Wqh, bq, qf, qh);
    qkv_gemm<<<grd, blk, 0, stream>>>(xh, Wkh, bk, kf, kh);
    qkv_gemm<<<grd, blk, 0, stream>>>(xh, Wvh, bv, vf, vh);
  }

  // 3) sampled-score sparsity measure (one wave per query row)
  measure_kernel<<<dim3(LL/8, BHH), 256, 0, stream>>>(qf, kf, index_key, meas);

  // 4) top-250 active queries per (b,h)
  topk_kernel<<<BHH, 512, 0, stream>>>(meas, idxq);

  // 5) sim = q_sel @ k^T / 6 (WMMA, key-blocked x3): 21 groups x 16 q-tiles
  sim_gemm<<<dim3(6, QPAD/16, BHH), dim3(32, 4), 0, stream>>>(qh, kh, idxq, simb);

  // 6) softmax rows -> attn (f16, zero-padded)
  softmax_kernel<<<dim3(QPAD, BHH), 128, 0, stream>>>(simb, attn);

  // 7) out_sel = attn @ v (WMMA, all 3 N-tiles per wave)
  outsel_gemm<<<dim3((QPAD/16)/4, BHH), dim3(32, 4), 0, stream>>>(attn, vh, outsel);

  // 8) lazy-query output = mean(V); then scatter active outputs
  vmean_kernel<<<BHH, 64, 0, stream>>>(vf, vmean);
  fill_out_kernel<<<dim3(LL, BHH), 64, 0, stream>>>(vmean, attnout);
  scatter_kernel<<<dim3(NQQ, BHH), 64, 0, stream>>>(idxq, outsel, attnout);

  // 9) residual + LN1 -> h1 (f32 + f16)
  ln1_kernel<<<MTOT, DD, 0, stream>>>(x, attnout, ln1g, ln1b, h1, h1h);

  // 10) FFN: gelu(h1 @ W1^T + b1) -> f16 ; gelu(f @ W2^T + b2) -> f32
  ffn_gemm<true ><<<dim3((MTOT/16 + 3)/4, DFFF/32), dim3(32, 4), 0, stream>>>(h1h, DD,   W1h, b1, (void*)f1h, DFFF);
  ffn_gemm<false><<<dim3((MTOT/16 + 3)/4, DD/32),   dim3(32, 4), 0, stream>>>(f1h, DFFF, W2h, b2, (void*)g2,  DD);

  // 11) residual + LN2 -> output
  ln2_kernel<<<MTOT, DD, 0, stream>>>(g2, h1, ln2g, ln2b, out);
}
